// Block_69578470195514
// MI455X (gfx1250) — matrix operand order repaired and checked by reading, not compiled, not run on hardware
//
#include <hip/hip_runtime.h>
#include <hip/hip_bf16.h>
#include <math.h>

// ---------------------------------------------------------------------------
// Transformer block for MI455X (gfx1250, wave32, WMMA bf16).
// B=8 T=1024 D=1024 H=16 DH=64. All GEMMs via v_wmma_f32_16x16x32_bf16.
// Column-axis softmax handled by a two-pass online-stats + recompute scheme.
// GEMM staging uses gfx1250 async global->LDS DMA (ASYNCcnt) with double
// buffering when the toolchain exposes the builtins; falls back to sync
// loads + ds_store otherwise.
// ---------------------------------------------------------------------------

#define Bc   8
#define Tc   1024
#define Dc   1024
#define Hc   16
#define DHc  64
#define BTc  (Bc * Tc)
#define BHc  (Bc * Hc)
#define FFc  (4 * Dc)
#define SCALE 0.03125f   // D^-0.5
#define EPS   1e-3f

typedef __bf16 bf16;
typedef __attribute__((ext_vector_type(16))) __bf16 v16bf;
typedef __attribute__((ext_vector_type(8)))  float  v8f;
typedef __attribute__((ext_vector_type(4)))  int    v4i;

#if defined(__AMDGCN__) &&                                                     \
    __has_builtin(__builtin_amdgcn_global_load_async_to_lds_b128) &&           \
    __has_builtin(__builtin_amdgcn_s_wait_asynccnt)
#define USE_ASYNC_LDS 1
#else
#define USE_ASYNC_LDS 0
#endif

#if USE_ASYNC_LDS
// 16B per-lane async DMA: global -> LDS, tracked with ASYNCcnt.
// Builtin signature: (int4 addrspace(1)*, int4 addrspace(3)*, imm off, imm cpol)
__device__ __forceinline__ void async_copy16(const bf16* g, bf16* l) {
  __builtin_amdgcn_global_load_async_to_lds_b128(
      (__attribute__((address_space(1))) v4i*)(void*)g,
      (__attribute__((address_space(3))) v4i*)(void*)l, 0, 0);
}
#endif

__device__ __forceinline__ v8f wmma_bf16(v16bf a, v16bf b, v8f c) {
  // (neg_a, A, neg_b, B, c_mod, C, reuse_a, reuse_b)
  return __builtin_amdgcn_wmma_f32_16x16x32_bf16(false, a, false, b, (short)0, c,
                                                 false, false);
}

// A-matrix fragment (16x32 bf16, row-major source, leading dim ld):
// lane L: m = L&15, hi = L>>4; slot j -> k = j + 16*hi
__device__ __forceinline__ v16bf load_a_frag(const bf16* tb, int ld, int lane) {
  const int m = lane & 15, hi = lane >> 4;
  const bf16* p0 = tb + (size_t)m * ld + 16 * hi;
  const bf16* p1 = tb + (size_t)m * ld + 16 * hi + 8;
  v16bf a;
#pragma unroll
  for (int j = 0; j < 8; ++j) { a[j] = p0[j]; a[j + 8] = p1[j]; }
  return a;
}

// B-matrix fragment (32x16 bf16) sourced from a [N][K] (column-major-of-B) tile:
// lane L: n = L&15, hi = L>>4; slot j -> k = j + 16*hi
__device__ __forceinline__ v16bf load_b_frag(const bf16* tb, int ld, int lane) {
  const int n = lane & 15, hi = lane >> 4;
  const bf16* p = tb + (size_t)n * ld + 16 * hi;
  v16bf b;
#pragma unroll
  for (int j = 0; j < 16; ++j) b[j] = p[j];
  return b;
}

// ---------------------------------------------------------------------------
// Weight conversion kernels (fp32 -> bf16, pre-transposed to [N][K] layout)
// ---------------------------------------------------------------------------
__global__ __launch_bounds__(256) void conv_qkv_kernel(
    const float* __restrict__ Wq, const float* __restrict__ Wk,
    const float* __restrict__ Wv, bf16* __restrict__ out) {
  int i = blockIdx.x * 256 + threadIdx.x;           // over 3*D*D
  if (i >= 3 * Dc * Dc) return;
  int row = i / Dc;                                  // j in [0,3D)
  int d   = i % Dc;
  int r   = row / Dc;                                // 0=q 1=k 2=v
  int jj  = row % Dc;
  int hh  = jj >> 6, e = jj & (DHc - 1);
  const float* W = (r == 0) ? Wq : (r == 1) ? Wk : Wv;    // [H][D][DH]
  out[i] = (bf16)W[((size_t)hh * Dc + d) * DHc + e];
}

__global__ __launch_bounds__(256) void transpose_conv_kernel(
    const float* __restrict__ in, bf16* __restrict__ out, int Kd, int Nd) {
  int i = blockIdx.x * 256 + threadIdx.x;           // over Kd*Nd (output-linear)
  if (i >= Kd * Nd) return;
  int n = i / Kd, kk = i % Kd;
  out[i] = (bf16)in[(size_t)kk * Nd + n];           // out[n][k] = in[k][n]
}

// ---------------------------------------------------------------------------
// LayerNorm: one 256-thread block per row of D=1024, fp32 math, bf16 out.
// ---------------------------------------------------------------------------
__global__ __launch_bounds__(256) void ln_kernel(
    const float* __restrict__ x, const float* __restrict__ g,
    const float* __restrict__ bta, bf16* __restrict__ out) {
  __shared__ float ws[8], wss[8];
  const int row = blockIdx.x, tid = threadIdx.x;
  const float* xr = x + (size_t)row * Dc;
  float s = 0.f, ss = 0.f;
  for (int i = tid; i < Dc; i += 256) { float v = xr[i]; s += v; ss += v * v; }
#pragma unroll
  for (int off = 16; off; off >>= 1) {
    s  += __shfl_xor(s, off);
    ss += __shfl_xor(ss, off);
  }
  if ((tid & 31) == 0) { ws[tid >> 5] = s; wss[tid >> 5] = ss; }
  __syncthreads();
  float st = 0.f, sst = 0.f;
#pragma unroll
  for (int w = 0; w < 8; ++w) { st += ws[w]; sst += wss[w]; }
  const float mu = st * (1.0f / Dc);
  const float var = sst * (1.0f / Dc) - mu * mu;
  const float rs = rsqrtf(var + EPS);
  bf16* orow = out + (size_t)row * Dc;
  for (int i = tid; i < Dc; i += 256)
    orow[i] = (bf16)((xr[i] - mu) * rs * g[i] + bta[i]);
}

// ---------------------------------------------------------------------------
// Generic tiled WMMA GEMM: C[M,N] = A[M,K](bf16) * Bt[N,K](bf16)^T + epilogue.
// Block 256 thr (8 waves), tile 128x128x32. Wave = 64x32 region (4x2 tiles).
// Double-buffered LDS; async DMA staging when available.
// mode 0: QKV scatter (q,k row-major per head; v transposed [e][t])
// mode 1: outF = resid + acc + bias          (proj residual, fp32 out)
// mode 2: outB = bf16(relu(acc + bias))      (FFN1)
// mode 3: outF = resid + acc + bias          (FFN2 final, resid==outF ok)
// ---------------------------------------------------------------------------
#define GBM 128
#define GBN 128
#define GBK 32

__global__ __launch_bounds__(256) void gemm_kernel(
    const bf16* __restrict__ A, const bf16* __restrict__ Bt,
    int M, int N, int K, int mode,
    const float* __restrict__ bias, const float* __restrict__ resid,
    float* __restrict__ outF, bf16* __restrict__ outB,
    bf16* __restrict__ qo, bf16* __restrict__ ko, bf16* __restrict__ vto) {
  __shared__ __align__(16) bf16 sA[2][GBM * GBK];
  __shared__ __align__(16) bf16 sB[2][GBN * GBK];

  const int tid = threadIdx.x, lane = tid & 31, wid = tid >> 5;
  const int n0 = blockIdx.x * GBN, m0 = blockIdx.y * GBM;
  const int nl = lane & 15, hi = lane >> 4;
  const int wm = wid >> 2, wn = wid & 3;      // 2x4 wave grid

  v8f acc[4][2];
#pragma unroll
  for (int i = 0; i < 4; ++i)
#pragma unroll
    for (int j = 0; j < 2; ++j)
#pragma unroll
      for (int r = 0; r < 8; ++r) acc[i][j][r] = 0.f;

  const int ldrow = tid >> 1;
  const int ldcol = (tid & 1) * 16;
  const bf16* gA = A  + (size_t)(m0 + ldrow) * K + ldcol;
  const bf16* gB = Bt + (size_t)(n0 + ldrow) * K + ldcol;
  const int ldsoff = ldrow * GBK + ldcol;

#if USE_ASYNC_LDS
  // Prologue: DMA tile k0=0 into buffer 0 (4 x 16B per thread).
  async_copy16(gA + 0, &sA[0][ldsoff]);
  async_copy16(gA + 8, &sA[0][ldsoff + 8]);
  async_copy16(gB + 0, &sB[0][ldsoff]);
  async_copy16(gB + 8, &sB[0][ldsoff + 8]);
#endif

  for (int k0 = 0; k0 < K; k0 += GBK) {
    const int cur = (k0 / GBK) & 1;
#if USE_ASYNC_LDS
    if (k0 + GBK < K) {
      const int kn = k0 + GBK;
      async_copy16(gA + kn + 0, &sA[cur ^ 1][ldsoff]);
      async_copy16(gA + kn + 8, &sA[cur ^ 1][ldsoff + 8]);
      async_copy16(gB + kn + 0, &sB[cur ^ 1][ldsoff]);
      async_copy16(gB + kn + 8, &sB[cur ^ 1][ldsoff + 8]);
      // Async ops complete in order: <=4 outstanding => current tile landed.
      __builtin_amdgcn_s_wait_asynccnt(4);
    } else {
      __builtin_amdgcn_s_wait_asynccnt(0);
    }
#else
    {
      const uint4* sa = (const uint4*)(gA + k0);
      *(uint4*)&sA[cur][ldsoff]     = sa[0];
      *(uint4*)&sA[cur][ldsoff + 8] = sa[1];
      const uint4* sb = (const uint4*)(gB + k0);
      *(uint4*)&sB[cur][ldsoff]     = sb[0];
      *(uint4*)&sB[cur][ldsoff + 8] = sb[1];
    }
    if (k0 + GBK < K) {   // gfx1250 global_prefetch_b8 of next K tiles
      __builtin_prefetch(gA + k0 + GBK, 0, 1);
      __builtin_prefetch(gB + k0 + GBK, 0, 1);
    }
#endif
    __syncthreads();

    v16bf af[4], bfv[2];
#pragma unroll
    for (int i = 0; i < 4; ++i)
      af[i] = load_a_frag(&sA[cur][(wm * 64 + i * 16) * GBK], GBK, lane);
#pragma unroll
    for (int j = 0; j < 2; ++j)
      bfv[j] = load_b_frag(&sB[cur][(wn * 32 + j * 16) * GBK], GBK, lane);
#pragma unroll
    for (int i = 0; i < 4; ++i)
#pragma unroll
      for (int j = 0; j < 2; ++j)
        acc[i][j] = wmma_bf16(af[i], bfv[j], acc[i][j]);
    __syncthreads();
  }

  // Epilogue
#pragma unroll
  for (int i = 0; i < 4; ++i)
#pragma unroll
    for (int j = 0; j < 2; ++j)
#pragma unroll
      for (int r = 0; r < 8; ++r) {
        const int m = m0 + wm * 64 + i * 16 + r + 8 * hi;
        const int n = n0 + wn * 32 + j * 16 + nl;
        const float v = acc[i][j][r];
        if (mode == 0) {
          const int reg = n >> 10, jj = n & (Dc - 1);
          const int hh = jj >> 6, e = jj & (DHc - 1);
          const int b = m >> 10, t = m & (Tc - 1);
          const size_t bh = (size_t)(b * Hc + hh);
          const bf16 bv = (bf16)v;
          if (reg == 0)      qo[(bh * Tc + t) * DHc + e] = bv;
          else if (reg == 1) ko[(bh * Tc + t) * DHc + e] = bv;
          else               vto[(bh * DHc + e) * Tc + t] = bv;
        } else if (mode == 2) {
          const float sv = v + bias[n];
          outB[(size_t)m * N + n] = (bf16)(sv > 0.f ? sv : 0.f);
        } else {  // 1 or 3
          outF[(size_t)m * N + n] = resid[(size_t)m * N + n] + v + bias[n];
        }
      }
}

// ---------------------------------------------------------------------------
// Attention pass 1: per-column (query-axis) online softmax stats.
// One wave per (b,h, 16-column block). S tiles via WMMA; columns combined
// across the two lane halves with shfl_xor(16).
// ---------------------------------------------------------------------------
__global__ __launch_bounds__(32) void att_stats_kernel(
    const bf16* __restrict__ q, const bf16* __restrict__ k,
    float* __restrict__ Mout, float* __restrict__ iZout) {
  const int lane = threadIdx.x;
  const int gid = blockIdx.x;
  const int bh = gid / (Tc / 16);
  const int s0 = (gid % (Tc / 16)) * 16;
  const bf16* qb = q + (size_t)bh * Tc * DHc;
  const bf16* kb = k + (size_t)bh * Tc * DHc;
  const int nl = lane & 15, hi = lane >> 4;
  const int s_lane = s0 + nl;

  // Columns fixed for this wave: preload both K fragments (k-dim = e, 2 steps).
  v16bf bf0 = load_b_frag(kb + (size_t)s0 * DHc + 0,  DHc, lane);
  v16bf bf1 = load_b_frag(kb + (size_t)s0 * DHc + 32, DHc, lane);

  float m = -INFINITY, z = 0.f;
  for (int t0 = s0; t0 < Tc; t0 += 16) {
    v8f cc;
#pragma unroll
    for (int r = 0; r < 8; ++r) cc[r] = 0.f;
    v16bf a0 = load_a_frag(qb + (size_t)t0 * DHc + 0,  DHc, lane);
    v16bf a1 = load_a_frag(qb + (size_t)t0 * DHc + 32, DHc, lane);
    cc = wmma_bf16(a0, bf0, cc);
    cc = wmma_bf16(a1, bf1, cc);

    float vals[8];
    float tm = -INFINITY;
#pragma unroll
    for (int r = 0; r < 8; ++r) {
      const int t = t0 + r + 8 * hi;
      float v = cc[r] * SCALE;
      if (t < s_lane) v = -INFINITY;          // causal: keep t >= s
      vals[r] = v;
      tm = fmaxf(tm, v);
    }
    tm = fmaxf(tm, __shfl_xor(tm, 16));       // merge the two lane halves
    const float nm = fmaxf(m, tm);            // finite after first tile (diag)
    float tz = 0.f;
#pragma unroll
    for (int r = 0; r < 8; ++r) tz += __expf(vals[r] - nm);
    tz += __shfl_xor(tz, 16);
    z = z * __expf(m - nm) + tz;
    m = nm;
  }
  if (hi == 0) {
    Mout[(size_t)bh * Tc + s_lane] = m;
    iZout[(size_t)bh * Tc + s_lane] = 1.0f / z;
  }
}

// ---------------------------------------------------------------------------
// Attention pass 2: recompute scores, P = exp(S-M)*invZ (bf16 via LDS),
// then att = P @ V with WMMA (V pre-transposed so B-frags are contiguous).
// One 256-thread block per (b,h, 64-row block).
// ---------------------------------------------------------------------------
#define ABM 64
__global__ __launch_bounds__(256) void att_apply_kernel(
    const bf16* __restrict__ q, const bf16* __restrict__ k,
    const bf16* __restrict__ vT, const float* __restrict__ Msm,
    const float* __restrict__ iZ, bf16* __restrict__ attb) {
  __shared__ __align__(16) bf16 sP[ABM * 32];

  const int tid = threadIdx.x, lane = tid & 31, wid = tid >> 5;
  const int gid = blockIdx.x;
  const int bh = gid / (Tc / ABM);
  const int t0 = (gid % (Tc / ABM)) * ABM;
  const int b = bh / Hc, hh = bh % Hc;
  const bf16* qb = q + (size_t)bh * Tc * DHc;
  const bf16* kb = k + (size_t)bh * Tc * DHc;
  const bf16* vb = vT + (size_t)bh * DHc * Tc;
  const int nl = lane & 15, hi = lane >> 4;

  // S-phase job: wave computes P rows [rt*16,+16) x cols [ch*16,+16) of the
  // 64x32 block.  Q fragments are invariant across the s-loop: preload.
  const int rt = wid >> 1, ch = wid & 1;
  v16bf qa0 = load_a_frag(qb + (size_t)(t0 + rt * 16) * DHc + 0,  DHc, lane);
  v16bf qa1 = load_a_frag(qb + (size_t)(t0 + rt * 16) * DHc + 32, DHc, lane);

  // AV-phase job: wave owns output tiles (rt2, cp*2+u).
  const int rt2 = wid & 3, cp = wid >> 2;
  v8f acc[2];
#pragma unroll
  for (int u = 0; u < 2; ++u)
#pragma unroll
    for (int r = 0; r < 8; ++r) acc[u][r] = 0.f;

  const int send = t0 + ABM;                 // causal: s <= max t in block
  for (int sb = 0; sb < send; sb += 32) {
    // ---- scores for this 64x32 block ----
    const int scol = sb + ch * 16;
    v8f cc;
#pragma unroll
    for (int r = 0; r < 8; ++r) cc[r] = 0.f;
    v16bf kf0 = load_b_frag(kb + (size_t)scol * DHc + 0,  DHc, lane);
    v16bf kf1 = load_b_frag(kb + (size_t)scol * DHc + 32, DHc, lane);
    cc = wmma_bf16(qa0, kf0, cc);
    cc = wmma_bf16(qa1, kf1, cc);

    const int sL = scol + nl;                // this lane's global column
    const float Mv = Msm[(size_t)bh * Tc + sL];
    const float zi = iZ[(size_t)bh * Tc + sL];
#pragma unroll
    for (int r = 0; r < 8; ++r) {
      const int t = t0 + rt * 16 + r + 8 * hi;
      const float v = cc[r] * SCALE;
      const float p = (t >= sL) ? __expf(v - Mv) * zi : 0.f;
      sP[(rt * 16 + r + 8 * hi) * 32 + ch * 16 + nl] = (bf16)p;
    }
    __syncthreads();

    // ---- att += P @ V  (K-dim = s, 32 per step) ----
    v16bf pa = load_a_frag(&sP[(rt2 * 16) * 32], 32, lane);
#pragma unroll
    for (int u = 0; u < 2; ++u) {
      const int ct = cp * 2 + u;
      v16bf vf = load_b_frag(vb + (size_t)(ct * 16) * Tc + sb, Tc, lane);
      acc[u] = wmma_bf16(pa, vf, acc[u]);
    }
    __syncthreads();
  }

  // concat-head layout: attb[(b*T + t)*D + h*DH + e]
#pragma unroll
  for (int u = 0; u < 2; ++u) {
    const int ct = cp * 2 + u;
#pragma unroll
    for (int r = 0; r < 8; ++r) {
      const int t = t0 + rt2 * 16 + r + 8 * hi;
      attb[((size_t)(b * Tc + t)) * Dc + hh * DHc + ct * 16 + nl] =
          (bf16)acc[u][r];
    }
  }
}

// ---------------------------------------------------------------------------
// Host-side orchestration
// ---------------------------------------------------------------------------
static char* carve(char*& p, size_t bytes) {
  char* r = p;
  p += (bytes + 255) & ~(size_t)255;
  return r;
}

extern "C" void kernel_launch(void* const* d_in, const int* in_sizes, int n_in,
                              void* d_out, int out_size, void* d_ws,
                              size_t ws_size, hipStream_t stream) {
  const float* x      = (const float*)d_in[0];
  const float* gamma1 = (const float*)d_in[1];
  const float* beta1  = (const float*)d_in[2];
  const float* Wq     = (const float*)d_in[3];
  const float* Wk     = (const float*)d_in[4];
  const float* Wv     = (const float*)d_in[5];
  const float* Wp     = (const float*)d_in[6];
  const float* bp     = (const float*)d_in[7];
  const float* gamma2 = (const float*)d_in[8];
  const float* beta2  = (const float*)d_in[9];
  const float* W1     = (const float*)d_in[10];
  const float* b1     = (const float*)d_in[11];
  const float* W2     = (const float*)d_in[12];
  const float* b2     = (const float*)d_in[13];
  float* out = (float*)d_out;

  char* p = (char*)d_ws;
  bf16* xn1    = (bf16*)carve(p, (size_t)BTc * Dc * 2);
  bf16* wqkv_t = (bf16*)carve(p, (size_t)3 * Dc * Dc * 2);
  bf16* wp_t   = (bf16*)carve(p, (size_t)Dc * Dc * 2);
  bf16* w1_t   = (bf16*)carve(p, (size_t)Dc * FFc * 2);
  bf16* w2_t   = (bf16*)carve(p, (size_t)FFc * Dc * 2);
  bf16* qb     = (bf16*)carve(p, (size_t)BHc * Tc * DHc * 2);
  bf16* kb     = (bf16*)carve(p, (size_t)BHc * Tc * DHc * 2);
  bf16* vTb    = (bf16*)carve(p, (size_t)BHc * DHc * Tc * 2);
  float* Msm   = (float*)carve(p, (size_t)BHc * Tc * 4);
  float* iZ    = (float*)carve(p, (size_t)BHc * Tc * 4);
  bf16* attb   = (bf16*)carve(p, (size_t)BTc * Dc * 2);
  bf16* xn2    = (bf16*)carve(p, (size_t)BTc * Dc * 2);
  bf16* h1     = (bf16*)carve(p, (size_t)BTc * FFc * 2);

  // 1) Weight conversions (fp32 -> bf16, [N][K] layouts).
  conv_qkv_kernel<<<(3 * Dc * Dc) / 256, 256, 0, stream>>>(Wq, Wk, Wv, wqkv_t);
  transpose_conv_kernel<<<(Dc * Dc) / 256, 256, 0, stream>>>(Wp, wp_t, Dc, Dc);
  transpose_conv_kernel<<<(Dc * FFc) / 256, 256, 0, stream>>>(W1, w1_t, Dc, FFc);
  transpose_conv_kernel<<<(FFc * Dc) / 256, 256, 0, stream>>>(W2, w2_t, FFc, Dc);

  // 2) LN1
  ln_kernel<<<BTc, 256, 0, stream>>>(x, gamma1, beta1, xn1);

  // 3) QKV projection: [BT,3D] = xn1 @ Wqkv, scattered to q/k/vT.
  gemm_kernel<<<dim3(3 * Dc / GBN, BTc / GBM), 256, 0, stream>>>(
      xn1, wqkv_t, BTc, 3 * Dc, Dc, /*mode=*/0, nullptr, nullptr, nullptr,
      nullptr, qb, kb, vTb);

  // 4) Column-softmax stats, then attention apply.
  att_stats_kernel<<<BHc * (Tc / 16), 32, 0, stream>>>(qb, kb, Msm, iZ);
  att_apply_kernel<<<BHc * (Tc / ABM), 256, 0, stream>>>(qb, kb, vTb, Msm, iZ,
                                                         attb);

  // 5) Output projection + residual -> x1 (fp32, lives in d_out).
  gemm_kernel<<<dim3(Dc / GBN, BTc / GBM), 256, 0, stream>>>(
      attb, wp_t, BTc, Dc, Dc, /*mode=*/1, bp, x, out, nullptr, nullptr,
      nullptr, nullptr);

  // 6) LN2 on x1.
  ln_kernel<<<BTc, 256, 0, stream>>>(out, gamma2, beta2, xn2);

  // 7) FFN1: h1 = relu(xn2 @ W1 + b1) (bf16).
  gemm_kernel<<<dim3(FFc / GBN, BTc / GBM), 256, 0, stream>>>(
      xn2, w1_t, BTc, FFc, Dc, /*mode=*/2, b1, nullptr, nullptr, h1, nullptr,
      nullptr, nullptr);

  // 8) FFN2: out = x1 + h1 @ W2 + b2 (read-modify-write d_out, disjoint elems).
  gemm_kernel<<<dim3(Dc / GBN, BTc / GBM), 256, 0, stream>>>(
      h1, w2_t, BTc, Dc, FFc, /*mode=*/3, b2, out, out, nullptr, nullptr,
      nullptr, nullptr);

  (void)in_sizes; (void)n_in; (void)out_size; (void)ws_size;
}
